// TorchL2ROneLayerEncDecGruSeqPredWithPOS_75179107549685
// MI455X (gfx1250) — compile-verified, matrix operand order checked by reading, not run on hardware
//
#include <hip/hip_runtime.h>

// ---------------- model dimensions ----------------
#define BTOT   2048
#define SEQ    64
#define DEMB   256     // D_EC == D_DE
#define DEP    64
#define PLIN   128
#define PFIN   128
#define HENC   512
#define HDEC   640
#define CLIN   512
#define NCLS   128
#define STEPS  63      // MAX_STEPS-1

typedef __attribute__((ext_vector_type(16))) __bf16 v16bf;
typedef __attribute__((ext_vector_type(8)))  float  v8f;

__device__ __forceinline__ v8f wmma_bf16(v16bf a, v16bf b, v8f c) {
  // D = A(16x32 bf16) * B(32x16 bf16) + C(16x16 f32)
  return __builtin_amdgcn_wmma_f32_16x16x32_bf16(false, a, false, b, (short)0, c,
                                                 false, false);
}

// Load the 16 bf16 values this lane contributes to a 16x16x32 WMMA A or B
// operand, from a row-major bf16 row pointer (A: row = batch row lane&15,
// B: row = weight output row for column lane&15).
// K pattern per ISA 7.12.2: lane half 0 -> K {0..7, 16..23}, half 1 -> +8.
// Lowers to 2x b128 loads (global or ds).
__device__ __forceinline__ v16bf frag_bf16(const __bf16* __restrict__ row, int k0, int half) {
  union { v16bf v; unsigned int u[8]; } f;
  const unsigned int* p = (const unsigned int*)(row + k0 + half * 8);
#pragma unroll
  for (int q = 0; q < 4; ++q) { f.u[q] = p[q]; f.u[q + 4] = p[q + 8]; }
  return f.v;
}

__device__ __forceinline__ float gelu_f(float x) {
  return 0.5f * x * (1.0f + erff(x * 0.70710678118654752f));   // exact erf GELU
}
__device__ __forceinline__ float sigm_f(float x) { return 1.0f / (1.0f + expf(-x)); }

// ---------------- weight conversion f32 -> bf16 ----------------
__global__ void cvt_f32_bf16(const float* __restrict__ s, __bf16* __restrict__ d, int n) {
  int i = blockIdx.x * blockDim.x + threadIdx.x;
  if (i < n) d[i] = (__bf16)s[i];
}

// ---------------- pos MLP: dec_h[:, 0:128] = gelu(gelu(E_pos[pos]@Wp1^T+b)@Wp2^T+b)
__global__ void __launch_bounds__(32)
pos_kernel(const int* __restrict__ pos_ids, const __bf16* __restrict__ Epos,
           const __bf16* __restrict__ Wp1, const float* __restrict__ bp1,
           const __bf16* __restrict__ Wp2, const float* __restrict__ bp2,
           float* __restrict__ dec_h) {
  __shared__ __bf16 h1[16][PLIN];
  const int lane = threadIdx.x & 31;
  const int half = lane >> 4, mlo = lane & 15;
  const int r0 = blockIdx.x * 16;

  const __bf16* xrow = Epos + (size_t)pos_ids[r0 + mlo] * DEP;
#pragma unroll
  for (int nt = 0; nt < PLIN / 16; ++nt) {
    const int ncol = nt * 16 + mlo;
    v8f acc; float bb = bp1[ncol];
#pragma unroll
    for (int v = 0; v < 8; ++v) acc[v] = bb;
#pragma unroll
    for (int k = 0; k < DEP; k += 32)
      acc = wmma_bf16(frag_bf16(xrow, k, half),
                      frag_bf16(Wp1 + (size_t)ncol * DEP, k, half), acc);
#pragma unroll
    for (int v = 0; v < 8; ++v) h1[v + 8 * half][ncol] = (__bf16)gelu_f(acc[v]);
  }
  __syncthreads();
#pragma unroll
  for (int nt = 0; nt < PFIN / 16; ++nt) {
    const int ncol = nt * 16 + mlo;
    v8f acc; float bb = bp2[ncol];
#pragma unroll
    for (int v = 0; v < 8; ++v) acc[v] = bb;
#pragma unroll
    for (int k = 0; k < PLIN; k += 32)
      acc = wmma_bf16(frag_bf16(&h1[mlo][0], k, half),
                      frag_bf16(Wp2 + (size_t)ncol * PLIN, k, half), acc);
#pragma unroll
    for (int v = 0; v < 8; ++v)
      dec_h[(size_t)(r0 + v + 8 * half) * HDEC + ncol] = gelu_f(acc[v]);
  }
}

// Triple-accumulator K-loop with software-pipelined (double-buffered) fragment
// loads: chunk k+1 is issued before the WMMAs of chunk k, hiding load latency.
#define GATE_KLOOP(KDIM, AROW, W0, W1, W2, ACC0, ACC1, ACC2)                     \
  {                                                                              \
    v16bf a0  = frag_bf16((AROW), 0, half);                                      \
    v16bf b00 = frag_bf16((W0), 0, half);                                        \
    v16bf b10 = frag_bf16((W1), 0, half);                                        \
    v16bf b20 = frag_bf16((W2), 0, half);                                        \
    _Pragma("unroll")                                                            \
    for (int k = 32; k < (KDIM); k += 32) {                                      \
      v16bf a1  = frag_bf16((AROW), k, half);                                    \
      v16bf b01 = frag_bf16((W0), k, half);                                      \
      v16bf b11 = frag_bf16((W1), k, half);                                      \
      v16bf b21 = frag_bf16((W2), k, half);                                      \
      ACC0 = wmma_bf16(a0, b00, ACC0);                                           \
      ACC1 = wmma_bf16(a0, b10, ACC1);                                           \
      ACC2 = wmma_bf16(a0, b20, ACC2);                                           \
      a0 = a1; b00 = b01; b10 = b11; b20 = b21;                                  \
    }                                                                            \
    ACC0 = wmma_bf16(a0, b00, ACC0);                                             \
    ACC1 = wmma_bf16(a0, b10, ACC1);                                             \
    ACC2 = wmma_bf16(a0, b20, ACC2);                                             \
  }

// Single-accumulator pipelined K-loop (classifier layers).
#define MLP_KLOOP(KDIM, AROW, W0, ACC)                                           \
  {                                                                              \
    v16bf a0 = frag_bf16((AROW), 0, half);                                       \
    v16bf b0 = frag_bf16((W0), 0, half);                                         \
    _Pragma("unroll")                                                            \
    for (int k = 32; k < (KDIM); k += 32) {                                      \
      v16bf a1 = frag_bf16((AROW), k, half);                                     \
      v16bf b1 = frag_bf16((W0), k, half);                                       \
      ACC = wmma_bf16(a0, b0, ACC);                                              \
      a0 = a1; b0 = b1;                                                          \
    }                                                                            \
    ACC = wmma_bf16(a0, b0, ACC);                                                \
  }

// ---------------- encoder GRU: 16-row batch tile per block, 64 steps in-kernel.
// f32 hidden state lives in REGISTERS (static tile->lane ownership); LDS holds
// only a bf16 mirror used as the WMMA A operand by all waves.
__global__ void __launch_bounds__(512)
enc_kernel(const int* __restrict__ chr_ids, const int* __restrict__ lens_g,
           const __bf16* __restrict__ Echr, const __bf16* __restrict__ Wih,
           const __bf16* __restrict__ Whh, const float* __restrict__ bih,
           const float* __restrict__ bhh, float* __restrict__ dec_h) {
  __shared__ __bf16 hb[16][HENC];   // 16 KB bf16 mirror of hidden state
  __shared__ int    tok[16];
  __shared__ int    lens[16];
  const int tid = threadIdx.x, lane = tid & 31, wv = tid >> 5;
  const int half = lane >> 4, mlo = lane & 15;
  const int r0 = blockIdx.x * 16;

  float hreg[2][8];                 // this lane's owned h elements (f32 master)
#pragma unroll
  for (int s = 0; s < 2; ++s)
#pragma unroll
    for (int v = 0; v < 8; ++v) hreg[s][v] = 0.0f;
  for (int i = tid; i < 16 * HENC; i += 512) ((__bf16*)hb)[i] = (__bf16)0.0f;
  if (tid < 16) lens[tid] = lens_g[r0 + tid];
  __syncthreads();

  for (int t = 0; t < SEQ; ++t) {
    if (tid < 16) tok[tid] = chr_ids[(r0 + tid) * SEQ + t];
    __syncthreads();
    const __bf16* xrow = Echr + (size_t)tok[mlo] * DEMB;

    float nh[2][8];
    int s = 0;
    for (int jt = wv; jt < HENC / 16; jt += 16, ++s) {   // 32 hidden tiles / 16 waves
      const int ncol = jt * 16 + mlo;
      v8f ar, az, ai, ah;
      const float br = bih[ncol] + bhh[ncol];
      const float bz = bih[HENC + ncol] + bhh[HENC + ncol];
      const float bi = bih[2 * HENC + ncol];
      const float bh = bhh[2 * HENC + ncol];
#pragma unroll
      for (int v = 0; v < 8; ++v) { ar[v] = br; az[v] = bz; ai[v] = bi; ah[v] = bh; }

      const __bf16* wr = Wih + (size_t)ncol * DEMB;
      const __bf16* wz = Wih + (size_t)(HENC + ncol) * DEMB;
      const __bf16* wn = Wih + (size_t)(2 * HENC + ncol) * DEMB;
      GATE_KLOOP(DEMB, xrow, wr, wz, wn, ar, az, ai)     // x contribution

      const __bf16* vr = Whh + (size_t)ncol * HENC;
      const __bf16* vz = Whh + (size_t)(HENC + ncol) * HENC;
      const __bf16* vn = Whh + (size_t)(2 * HENC + ncol) * HENC;
      GATE_KLOOP(HENC, &hb[mlo][0], vr, vz, vn, ar, az, ah)  // h contribution

#pragma unroll
      for (int v = 0; v < 8; ++v) {
        const float r  = sigm_f(ar[v]);
        const float z  = sigm_f(az[v]);
        const float nn = tanhf(ai[v] + r * ah[v]);
        nh[s][v] = (1.0f - z) * nn + z * hreg[s][v];
      }
    }
    __syncthreads();                                     // all reads of hb done
    s = 0;
    for (int jt = wv; jt < HENC / 16; jt += 16, ++s) {
      const int ncol = jt * 16 + mlo;
#pragma unroll
      for (int v = 0; v < 8; ++v) {
        const int m = v + 8 * half;
        hreg[s][v] = nh[s][v];
        hb[m][ncol] = (__bf16)nh[s][v];
        if (t == lens[m] - 1)                            // enc_take_last
          dec_h[(size_t)(r0 + m) * HDEC + PFIN + ncol] = nh[s][v];
      }
    }
    __syncthreads();
  }
}

// ---------------- decoder: GRUCell + classifier + argmax feedback, 63 steps
__global__ void __launch_bounds__(512)
dec_kernel(const __bf16* __restrict__ Edec, const __bf16* __restrict__ Wih,
           const __bf16* __restrict__ Whh, const float* __restrict__ bih,
           const float* __restrict__ bhh, const __bf16* __restrict__ Wc1,
           const float* __restrict__ bc1, const __bf16* __restrict__ Wc2,
           const float* __restrict__ bc2, const float* __restrict__ dec_h_init,
           float* __restrict__ out) {
  __shared__ __bf16 hb[16][HDEC];     // 20 KB bf16 mirror of hidden state
  __shared__ __bf16 c1[16][CLIN];     // 16 KB classifier hidden
  __shared__ float  wmaxv[8][16];
  __shared__ int    wmaxi[8][16];
  __shared__ int    cur[16];
  const int tid = threadIdx.x, lane = tid & 31, wv = tid >> 5;
  const int half = lane >> 4, mlo = lane & 15;
  const int r0 = blockIdx.x * 16;

  float hreg[3][8];                   // this lane's owned h elements (f32 master)
  {
    int s = 0;
    for (int jt = wv; jt < HDEC / 16; jt += 16, ++s) {
      const int ncol = jt * 16 + mlo;
#pragma unroll
      for (int v = 0; v < 8; ++v)
        hreg[s][v] = dec_h_init[(size_t)(r0 + v + 8 * half) * HDEC + ncol];
    }
  }
  for (int i = tid; i < 16 * HDEC; i += 512)
    ((__bf16*)hb)[i] = (__bf16)dec_h_init[(size_t)r0 * HDEC + i];
  if (tid < 16) cur[tid] = 1;                            // INIT_DEC_IDX
  __syncthreads();

  for (int t = 0; t < STEPS; ++t) {
    const __bf16* xrow = Edec + (size_t)cur[mlo] * DEMB;

    // ---- GRUCell gates (40 hidden tiles over 16 waves) ----
    float nh[3][8];
    int s = 0;
    for (int jt = wv; jt < HDEC / 16; jt += 16, ++s) {
      const int ncol = jt * 16 + mlo;
      v8f ar, az, ai, ah;
      const float br = bih[ncol] + bhh[ncol];
      const float bz = bih[HDEC + ncol] + bhh[HDEC + ncol];
      const float bi = bih[2 * HDEC + ncol];
      const float bh = bhh[2 * HDEC + ncol];
#pragma unroll
      for (int v = 0; v < 8; ++v) { ar[v] = br; az[v] = bz; ai[v] = bi; ah[v] = bh; }

      const __bf16* wr = Wih + (size_t)ncol * DEMB;
      const __bf16* wz = Wih + (size_t)(HDEC + ncol) * DEMB;
      const __bf16* wn = Wih + (size_t)(2 * HDEC + ncol) * DEMB;
      GATE_KLOOP(DEMB, xrow, wr, wz, wn, ar, az, ai)

      const __bf16* vr = Whh + (size_t)ncol * HDEC;
      const __bf16* vz = Whh + (size_t)(HDEC + ncol) * HDEC;
      const __bf16* vn = Whh + (size_t)(2 * HDEC + ncol) * HDEC;
      GATE_KLOOP(HDEC, &hb[mlo][0], vr, vz, vn, ar, az, ah)

#pragma unroll
      for (int v = 0; v < 8; ++v) {
        const float r  = sigm_f(ar[v]);
        const float z  = sigm_f(az[v]);
        const float nn = tanhf(ai[v] + r * ah[v]);
        nh[s][v] = (1.0f - z) * nn + z * hreg[s][v];
      }
    }
    __syncthreads();
    s = 0;
    for (int jt = wv; jt < HDEC / 16; jt += 16, ++s) {
      const int ncol = jt * 16 + mlo;
#pragma unroll
      for (int v = 0; v < 8; ++v) {
        hreg[s][v] = nh[s][v];
        hb[v + 8 * half][ncol] = (__bf16)nh[s][v];
      }
    }
    __syncthreads();

    // ---- classifier layer 1: gelu(h @ Wc1^T + bc1) -> c1 (bf16) ----
    for (int ct = wv; ct < CLIN / 16; ct += 16) {
      const int ncol = ct * 16 + mlo;
      v8f acc; float bb = bc1[ncol];
#pragma unroll
      for (int v = 0; v < 8; ++v) acc[v] = bb;
      const __bf16* wrow = Wc1 + (size_t)ncol * HDEC;
      MLP_KLOOP(HDEC, &hb[mlo][0], wrow, acc)
#pragma unroll
      for (int v = 0; v < 8; ++v) c1[v + 8 * half][ncol] = (__bf16)gelu_f(acc[v]);
    }
    __syncthreads();

    // ---- classifier layer 2 + per-row argmax (8 logit tiles on waves 0..7) ----
    if (wv < NCLS / 16) {
      const int ncol = wv * 16 + mlo;
      v8f acc; float bb = bc2[ncol];
#pragma unroll
      for (int v = 0; v < 8; ++v) acc[v] = bb;
      const __bf16* wrow = Wc2 + (size_t)ncol * CLIN;
      MLP_KLOOP(CLIN, &c1[mlo][0], wrow, acc)
#pragma unroll
      for (int v = 0; v < 8; ++v)
        out[(size_t)(r0 + v + 8 * half) * STEPS * NCLS + (size_t)t * NCLS + ncol] = acc[v];
      // per-wave argmax over this wave's 16 columns, for every batch row m
      for (int m = 0; m < 16; ++m) {
        float val = (half == (m >> 3)) ? acc[m & 7] : -3.4e38f;
        int   idx = ncol;
#pragma unroll
        for (int off = 16; off >= 1; off >>= 1) {
          float v2 = __shfl_xor(val, off, 32);
          int   i2 = __shfl_xor(idx, off, 32);
          if (v2 > val || (v2 == val && i2 < idx)) { val = v2; idx = i2; }
        }
        if (lane == 0) { wmaxv[wv][m] = val; wmaxi[wv][m] = idx; }
      }
    }
    __syncthreads();
    if (tid < 16) {                                      // combine 8 wave partials
      float bv = wmaxv[0][tid]; int bi2 = wmaxi[0][tid];
      for (int w2 = 1; w2 < 8; ++w2) {
        float v2 = wmaxv[w2][tid];
        if (v2 > bv || (v2 == bv && wmaxi[w2][tid] < bi2)) { bv = v2; bi2 = wmaxi[w2][tid]; }
      }
      cur[tid] = bi2;                                    // argmax feedback
    }
    __syncthreads();
  }
}

// ---------------- host launcher ----------------
extern "C" void kernel_launch(void* const* d_in, const int* in_sizes, int n_in,
                              void* d_out, int out_size, void* d_ws, size_t ws_size,
                              hipStream_t stream) {
  (void)in_sizes; (void)n_in; (void)out_size; (void)ws_size;
  const int*   chr_ids = (const int*)  d_in[0];
  const int*   pos_ids = (const int*)  d_in[1];
  const int*   lens    = (const int*)  d_in[2];
  const float* E_chr   = (const float*)d_in[3];
  const float* E_pos   = (const float*)d_in[4];
  const float* E_dec   = (const float*)d_in[5];
  const float* Wp1     = (const float*)d_in[6];
  const float* bp1     = (const float*)d_in[7];
  const float* Wp2     = (const float*)d_in[8];
  const float* bp2     = (const float*)d_in[9];
  const float* We_ih   = (const float*)d_in[10];
  const float* We_hh   = (const float*)d_in[11];
  const float* be_ih   = (const float*)d_in[12];
  const float* be_hh   = (const float*)d_in[13];
  const float* Wd_ih   = (const float*)d_in[14];
  const float* Wd_hh   = (const float*)d_in[15];
  const float* bd_ih   = (const float*)d_in[16];
  const float* bd_hh   = (const float*)d_in[17];
  const float* Wc1     = (const float*)d_in[18];
  const float* bc1     = (const float*)d_in[19];
  const float* Wc2     = (const float*)d_in[20];
  const float* bc2     = (const float*)d_in[21];

  // workspace layout: bf16 weight copies, then f32 dec_h [B, HDEC]
  __bf16* w = (__bf16*)d_ws;
  size_t  o = 0;
  auto take = [&](size_t n) { __bf16* p = w + o; o += n; return p; };
  __bf16* Echr_b = take((size_t)128 * DEMB);
  __bf16* Epos_b = take((size_t)128 * DEP);
  __bf16* Edec_b = take((size_t)128 * DEMB);
  __bf16* Wp1_b  = take((size_t)PLIN * DEP);
  __bf16* Wp2_b  = take((size_t)PFIN * PLIN);
  __bf16* Weih_b = take((size_t)3 * HENC * DEMB);
  __bf16* Wehh_b = take((size_t)3 * HENC * HENC);
  __bf16* Wdih_b = take((size_t)3 * HDEC * DEMB);
  __bf16* Wdhh_b = take((size_t)3 * HDEC * HDEC);
  __bf16* Wc1_b  = take((size_t)CLIN * HDEC);
  __bf16* Wc2_b  = take((size_t)NCLS * CLIN);
  if (o & 1) ++o;
  float* dec_h = (float*)(w + o);

  auto cvt = [&](const float* s, __bf16* d, int n) {
    cvt_f32_bf16<<<(n + 255) / 256, 256, 0, stream>>>(s, d, n);
  };
  cvt(E_chr, Echr_b, 128 * DEMB);
  cvt(E_pos, Epos_b, 128 * DEP);
  cvt(E_dec, Edec_b, 128 * DEMB);
  cvt(Wp1,   Wp1_b,  PLIN * DEP);
  cvt(Wp2,   Wp2_b,  PFIN * PLIN);
  cvt(We_ih, Weih_b, 3 * HENC * DEMB);
  cvt(We_hh, Wehh_b, 3 * HENC * HENC);
  cvt(Wd_ih, Wdih_b, 3 * HDEC * DEMB);
  cvt(Wd_hh, Wdhh_b, 3 * HDEC * HDEC);
  cvt(Wc1,   Wc1_b,  CLIN * HDEC);
  cvt(Wc2,   Wc2_b,  NCLS * CLIN);

  const int nblk = BTOT / 16;   // 128 batch tiles
  pos_kernel<<<nblk, 32, 0, stream>>>(pos_ids, Epos_b, Wp1_b, bp1, Wp2_b, bp2, dec_h);
  enc_kernel<<<nblk, 512, 0, stream>>>(chr_ids, lens, Echr_b, Weih_b, Wehh_b,
                                       be_ih, be_hh, dec_h);
  dec_kernel<<<nblk, 512, 0, stream>>>(Edec_b, Wdih_b, Wdhh_b, bd_ih, bd_hh,
                                       Wc1_b, bc1, Wc2_b, bc2, dec_h, (float*)d_out);
}
